// BasePointNet2_28673201668472
// MI455X (gfx1250) — matrix operand order repaired and checked by reading, compiled, not run on hardware
//
#include <hip/hip_runtime.h>
#include <hip/hip_bf16.h>
#include <stdint.h>
#include <stddef.h>

#define BB 4
#define BN_EPS 1e-5f

typedef __bf16 bf16_t;
typedef __attribute__((ext_vector_type(16))) __bf16 v16bf;
typedef __attribute__((ext_vector_type(8)))  float  v8f;
typedef __attribute__((ext_vector_type(4)))  unsigned int u32x4;

union V16 { u32x4 q[2]; v16bf v; };

static __device__ __forceinline__ unsigned short f2bfb(float f) {
  unsigned u = __builtin_bit_cast(unsigned, f);
  u += 0x7FFFu + ((u >> 16) & 1u);            // round-to-nearest-even
  return (unsigned short)(u >> 16);
}
static __device__ __forceinline__ bf16_t f2bf(float f) {
  unsigned short h = f2bfb(f);
  return __builtin_bit_cast(bf16_t, h);
}

static inline int al32(int x) { return (x + 31) & ~31; }

// ---------------------------------------------------------------------------
// Farthest point sampling: one block per batch, min-dist array in LDS.
// ---------------------------------------------------------------------------
__global__ void fps_kernel(const float* __restrict__ xyz, int N, int S,
                           int* __restrict__ fidx) {
  int b = blockIdx.x;
  const float* px = xyz + (size_t)b * N * 3;
  __shared__ float md[4096];
  __shared__ float sv[256];
  __shared__ int   si[256];
  for (int i = threadIdx.x; i < N; i += blockDim.x) md[i] = 1e10f;
  __syncthreads();
  int far = 0;
  for (int s = 0; s < S; ++s) {
    if (threadIdx.x == 0) fidx[b * S + s] = far;
    float cx = px[far * 3 + 0], cy = px[far * 3 + 1], cz = px[far * 3 + 2];
    float best = -1.0f; int bi = 0;
    for (int i = threadIdx.x; i < N; i += blockDim.x) {
      float dx = px[i * 3 + 0] - cx, dy = px[i * 3 + 1] - cy, dz = px[i * 3 + 2] - cz;
      float d = dx * dx + dy * dy + dz * dz;
      float m = fminf(md[i], d);
      md[i] = m;
      if (m > best) { best = m; bi = i; }
    }
    sv[threadIdx.x] = best; si[threadIdx.x] = bi;
    __syncthreads();
    for (int st = 128; st > 0; st >>= 1) {
      if (threadIdx.x < st && sv[threadIdx.x + st] > sv[threadIdx.x]) {
        sv[threadIdx.x] = sv[threadIdx.x + st];
        si[threadIdx.x] = si[threadIdx.x + st];
      }
      __syncthreads();
    }
    far = si[0];
    __syncthreads();
  }
}

__global__ void gather_xyz_kernel(const float* __restrict__ xyz, const int* __restrict__ fidx,
                                  int N, int S, float* __restrict__ nxyz) {
  int t = blockIdx.x * blockDim.x + threadIdx.x;
  if (t >= BB * S * 3) return;
  int c = t % 3; int bs = t / 3; int s = bs % S; int b = bs / S;
  nxyz[t] = xyz[((size_t)b * N + fidx[b * S + s]) * 3 + c];
}

// Ball query: first K points (index order) within radius; pad with first.
__global__ void ball_query_kernel(const float* __restrict__ xyz, const float* __restrict__ nxyz,
                                  int N, int S, int K, float r2, int* __restrict__ gidx) {
  int t = blockIdx.x * blockDim.x + threadIdx.x;
  if (t >= BB * S) return;
  int b = t / S;
  const float* px = xyz + (size_t)b * N * 3;
  float cx = nxyz[t * 3 + 0], cy = nxyz[t * 3 + 1], cz = nxyz[t * 3 + 2];
  int* out = gidx + (size_t)t * K;
  int cnt = 0, first = -1;
  for (int i = 0; i < N && cnt < K; ++i) {
    float dx = px[i * 3 + 0] - cx, dy = px[i * 3 + 1] - cy, dz = px[i * 3 + 2] - cz;
    float d = dx * dx + dy * dy + dz * dz;
    if (d <= r2) {
      if (first < 0) first = i;
      out[cnt++] = i;
    }
  }
  if (first < 0) first = 0;
  for (; cnt < K; ++cnt) out[cnt] = first;
}

// grouped features -> [B*S*K, CinP] bf16 (K-padded, pad columns zeroed)
__global__ void group_gather_kernel(const float* __restrict__ xyz, const float* __restrict__ nxyz,
                                    const float* __restrict__ points, int Cp,
                                    const int* __restrict__ gidx, int N, int S, int K,
                                    int CinP, bf16_t* __restrict__ out) {
  size_t t = (size_t)blockIdx.x * blockDim.x + threadIdx.x;
  int Cin = Cp + 3;
  size_t total = (size_t)BB * S * K * CinP;
  if (t >= total) return;
  int c = (int)(t % CinP);
  size_t row = t / CinP;
  if (c >= Cin) { out[t] = f2bf(0.0f); return; }
  size_t bs = row / K;
  int s = (int)(bs % S); int b = (int)(bs / S);
  int gi = gidx[row];
  float v;
  if (c < Cp) {
    v = points[((size_t)b * N + gi) * Cp + c];
  } else {
    int cc = c - Cp;
    v = xyz[((size_t)b * N + gi) * 3 + cc] - nxyz[((size_t)b * S + s) * 3 + cc];
  }
  out[t] = f2bf(v);
}

__global__ void zero2_kernel(float* __restrict__ a, float* __restrict__ b, int n) {
  int i = blockIdx.x * blockDim.x + threadIdx.x;
  if (i < n) { a[i] = 0.0f; b[i] = 0.0f; }
}

// pack f32 row-major W[Nout,Kd] -> zero-padded bf16 Wp[NP,KP] (once per layer)
__global__ void pack_w_kernel(const float* __restrict__ W, int Nout, int Kd, int KP,
                              bf16_t* __restrict__ Wp, long long total) {
  long long t = (long long)blockIdx.x * blockDim.x + threadIdx.x;
  if (t >= total) return;
  int c = (int)(t % KP);
  long long r = t / KP;
  float v = (r < Nout && c < Kd) ? W[r * Kd + c] : 0.0f;
  Wp[t] = f2bf(v);
}

// ---------------------------------------------------------------------------
// GEMM: out[M,Nout] = A[M,KP](bf16, K-padded) * Wp[NP,KP]^T(bf16, padded) + bias
// f32 accumulation via V_WMMA_F32_16X16X32_BF16. 8 waves, 128x32 macro tile,
// 2 accumulators/wave sharing one A fragment. Hot loop is guard-free: pure
// b128 global->LDS copies + b128 fragment loads + 2 WMMA.
// Epilogue adds bias and accumulates per-column sum/sumsq for batchnorm.
// ---------------------------------------------------------------------------
__global__ __launch_bounds__(256)
void gemm_bias_wmma(const bf16_t* __restrict__ A, int KP,
                    const bf16_t* __restrict__ Wp, const float* __restrict__ bias,
                    int M, int Nout,
                    float* __restrict__ out, float* __restrict__ cs, float* __restrict__ cs2) {
  constexpr int LDA = 40;                       // elements; 80B row stride (16B multiple)
  __shared__ __align__(16) bf16_t As[128 * LDA];
  __shared__ __align__(16) bf16_t Bs[2 * 32 * 16];  // per-lane fragment order, 2 n-tiles
  int mBase = blockIdx.x * 128;
  int nBase = blockIdx.y * 32;
  int tid = threadIdx.x;
  int wave = tid >> 5, lane = tid & 31;
  int hi = lane >> 4, ln = lane & 15;

  v8f acc0 = {}, acc1 = {};
  for (int k0 = 0; k0 < KP; k0 += 32) {
    // stage A tile 128x32 bf16: 512 b128 slots, 2 per thread, no guards
    {
      int t0 = tid;
#pragma unroll
      for (int it = 0; it < 2; ++it, t0 += 256) {
        int r = t0 >> 2, c8 = t0 & 3;
        const u32x4* src = (const u32x4*)(A + (size_t)(mBase + r) * KP + k0 + c8 * 8);
        *(u32x4*)(As + r * LDA + c8 * 8) = *src;
      }
    }
    // stage B 32x32 from padded bf16 weights: single b128 copy per thread
    if (tid < 128) {
      int nt = tid >> 6;                 // which 16-wide n-tile
      int lane_s = (tid >> 1) & 31;      // destination lane
      int e8 = tid & 1;                  // low/high 8 elements
      int hs = lane_s >> 4, ns = lane_s & 15;
      int gn = nBase + nt * 16 + ns;     // always < NP (padded)
      int kb = (hs << 3) + e8 * 16;      // fragment element -> K mapping
      *(u32x4*)(Bs + nt * 512 + lane_s * 16 + e8 * 8) =
          *(const u32x4*)(Wp + (size_t)gn * KP + k0 + kb);
    }
    if (k0 + 32 < KP)
      __builtin_prefetch(A + (size_t)(mBase + (tid & 127)) * KP + k0 + 32, 0, 1);
    __syncthreads();

    // A fragment: per 16-bit 16x32 layout, each lane = two contiguous 8-elem K runs
    int am = wave * 16 + ln;
    V16 ua, ub0, ub1;
    ua.q[0] = *(const u32x4*)(As + am * LDA + (hi << 3));
    ua.q[1] = *(const u32x4*)(As + am * LDA + (hi << 3) + 16);
    ub0.q[0] = *(const u32x4*)(Bs + lane * 16);
    ub0.q[1] = *(const u32x4*)(Bs + lane * 16 + 8);
    ub1.q[0] = *(const u32x4*)(Bs + 512 + lane * 16);
    ub1.q[1] = *(const u32x4*)(Bs + 512 + lane * 16 + 8);

    acc0 = __builtin_amdgcn_wmma_f32_16x16x32_bf16(false, ua.v, false, ub0.v,
                                                   (short)0, acc0, false, false);
    acc1 = __builtin_amdgcn_wmma_f32_16x16x32_bf16(false, ua.v, false, ub1.v,
                                                   (short)0, acc1, false, false);
    __syncthreads();
  }

  // epilogue: bias, store f32, per-column BN stats
  int m0 = mBase + wave * 16 + hi * 8;
#pragma unroll
  for (int t = 0; t < 2; ++t) {
    int n = nBase + t * 16 + ln;
    if (n < Nout) {
      float bv = bias[n];
      float s = 0.0f, ss = 0.0f;
      v8f a = (t == 0) ? acc0 : acc1;
#pragma unroll
      for (int r = 0; r < 8; ++r) {
        float v = a[r] + bv;
        out[(size_t)(m0 + r) * Nout + n] = v;
        s += v; ss += v * v;
      }
      atomicAdd(&cs[n], s);
      atomicAdd(&cs2[n], ss);
    }
  }
}

// batchnorm + relu; f32 in-place (stride C) + K-padded bf16 copy (stride CP)
__global__ void bn_relu_kernel(const float* __restrict__ x, int M, int C, int CP,
                               const float* __restrict__ cs, const float* __restrict__ cs2,
                               const float* __restrict__ g, const float* __restrict__ be,
                               float* __restrict__ yf, bf16_t* __restrict__ yb) {
  size_t t = (size_t)blockIdx.x * blockDim.x + threadIdx.x;
  size_t total = (size_t)M * CP;
  if (t >= total) return;
  int c = (int)(t % CP);
  size_t r = t / CP;
  if (c >= C) { yb[t] = f2bf(0.0f); return; }
  float inv = 1.0f / (float)M;
  float mean = cs[c] * inv;
  float var = cs2[c] * inv - mean * mean;
  float v = g[c] * (x[r * C + c] - mean) * rsqrtf(var + BN_EPS) + be[c];
  v = fmaxf(v, 0.0f);
  yf[r * C + c] = v;
  yb[t] = f2bf(v);
}

// max over K axis into concatenated output at column offset
__global__ void maxpool_kernel(const float* __restrict__ in, int S, int K, int C,
                               float* __restrict__ out, int Ctot, int coloff) {
  int t = blockIdx.x * blockDim.x + threadIdx.x;
  int total = BB * S * C;
  if (t >= total) return;
  int c = t % C; int bs = t / C;
  float m = -3.0e38f;
  const float* p = in + (size_t)bs * K * C + c;
  for (int k = 0; k < K; ++k) m = fmaxf(m, p[(size_t)k * C]);
  out[(size_t)bs * Ctot + coloff + c] = m;
}

// f32 -> bf16 copy into strided (padded) concat buffer
__global__ void skip_copy_kernel(const float* __restrict__ src, int M, int C,
                                 bf16_t* __restrict__ dst, int CtotP, int coloff) {
  size_t t = (size_t)blockIdx.x * blockDim.x + threadIdx.x;
  size_t total = (size_t)M * C;
  if (t >= total) return;
  int c = (int)(t % C);
  size_t r = t / C;
  dst[r * CtotP + coloff + c] = f2bf(src[t]);
}

// 3-NN inverse-distance interpolation; one block per (point, batch)
__global__ void interp_kernel(const float* __restrict__ xyz1, const float* __restrict__ xyz2,
                              const float* __restrict__ p2, int N1, int N2, int C2,
                              bf16_t* __restrict__ dst, int CtotP, int coloff) {
  int i = blockIdx.x, b = blockIdx.y;
  __shared__ int sj[3];
  __shared__ float sw[3];
  if (threadIdx.x == 0) {
    const float* q = xyz1 + ((size_t)b * N1 + i) * 3;
    float qx = q[0], qy = q[1], qz = q[2];
    float d0 = 3.0e38f, d1 = 3.0e38f, d2 = 3.0e38f;
    int j0 = 0, j1 = 0, j2 = 0;
    for (int j = 0; j < N2; ++j) {
      const float* p = xyz2 + ((size_t)b * N2 + j) * 3;
      float dx = p[0] - qx, dy = p[1] - qy, dz = p[2] - qz;
      float d = dx * dx + dy * dy + dz * dz;
      if (d < d0)      { d2 = d1; j2 = j1; d1 = d0; j1 = j0; d0 = d; j0 = j; }
      else if (d < d1) { d2 = d1; j2 = j1; d1 = d;  j1 = j; }
      else if (d < d2) { d2 = d;  j2 = j; }
    }
    float w0 = 1.0f / (d0 + 1e-8f), w1 = 1.0f / (d1 + 1e-8f), w2 = 1.0f / (d2 + 1e-8f);
    float wsum = w0 + w1 + w2;
    sj[0] = j0; sj[1] = j1; sj[2] = j2;
    sw[0] = w0 / wsum; sw[1] = w1 / wsum; sw[2] = w2 / wsum;
  }
  __syncthreads();
  size_t rb = ((size_t)b * N1 + i) * CtotP + coloff;
  const float* r0 = p2 + ((size_t)b * N2 + sj[0]) * C2;
  const float* r1 = p2 + ((size_t)b * N2 + sj[1]) * C2;
  const float* r2 = p2 + ((size_t)b * N2 + sj[2]) * C2;
  float w0 = sw[0], w1 = sw[1], w2 = sw[2];
  for (int c = threadIdx.x; c < C2; c += blockDim.x)
    dst[rb + c] = f2bf(w0 * r0[c] + w1 * r1[c] + w2 * r2[c]);
}

// (B, N, C) f32 -> (B, C, N) f32 final output
__global__ void transpose_out_kernel(const float* __restrict__ in, int N, int C,
                                     float* __restrict__ out) {
  size_t t = (size_t)blockIdx.x * blockDim.x + threadIdx.x;
  size_t total = (size_t)BB * N * C;
  if (t >= total) return;
  int c = (int)(t % C);
  size_t r = t / C;
  int n = (int)(r % N); int b = (int)(r / N);
  out[((size_t)b * C + c) * N + n] = in[t];
}

// ---------------------------------------------------------------------------
// Host orchestration
// ---------------------------------------------------------------------------
struct Mlp { const float *W, *b, *be, *g; int Cin, Cout; };
struct MlpOut { float* f; bf16_t* h; };

static void run_linear(hipStream_t st, const bf16_t* A, int KP, const Mlp& L, int M,
                       float* outF, bf16_t* wp, float* cs, float* cs2) {
  int NP = al32(L.Cout);
  long long wtot = (long long)NP * KP;
  pack_w_kernel<<<(unsigned)((wtot + 255) / 256), 256, 0, st>>>(L.W, L.Cout, L.Cin, KP, wp, wtot);
  zero2_kernel<<<(L.Cout + 255) / 256, 256, 0, st>>>(cs, cs2, L.Cout);
  dim3 grid((unsigned)(M / 128), (unsigned)(NP / 32));
  gemm_bias_wmma<<<grid, 256, 0, st>>>(A, KP, wp, L.b, M, L.Cout, outF, cs, cs2);
}

static MlpOut run_mlp(hipStream_t st, bf16_t* in0, int M, const Mlp* L, int nL,
                      char* const P[3], bf16_t* wp, float* cs, float* cs2) {
  int ii = 0;                 // callers place in0 (stride al32(L[0].Cin)) in P[0]
  const bf16_t* in = in0;
  int KP = al32(L[0].Cin);
  MlpOut mo = {nullptr, nullptr};
  for (int l = 0; l < nL; ++l) {
    int fi = (ii + 1) % 3, bi = (ii + 2) % 3;
    float* outF = (float*)P[fi];
    bf16_t* outB = (bf16_t*)P[bi];
    run_linear(st, in, KP, L[l], M, outF, wp, cs, cs2);
    int CP = al32(L[l].Cout);
    long long total = (long long)M * CP;
    bn_relu_kernel<<<(unsigned)((total + 255) / 256), 256, 0, st>>>(
        outF, M, L[l].Cout, CP, cs, cs2, L[l].g, L[l].be, outF, outB);
    in = outB; ii = bi; KP = CP;
    mo.f = outF; mo.h = outB;
  }
  return mo;
}

static void run_sa(hipStream_t st, const float* xyz, int N, const float* points, int Cp, int S,
                   float r0, int K0, const Mlp* br0, float r1, int K1, const Mlp* br1,
                   int* fidx, float* nxyz, float* out_p, int Ctot,
                   char* const P[3], bf16_t* wp, float* cs, float* cs2, int* gbuf) {
  fps_kernel<<<BB, 256, 0, st>>>(xyz, N, S, fidx);
  gather_xyz_kernel<<<(BB * S * 3 + 255) / 256, 256, 0, st>>>(xyz, fidx, N, S, nxyz);
  float rr[2] = {r0, r1}; int KK[2] = {K0, K1};
  const Mlp* brs[2] = {br0, br1};
  int coloff = 0;
  for (int br = 0; br < 2; ++br) {
    int K = KK[br];
    int M = BB * S * K;
    ball_query_kernel<<<(BB * S + 127) / 128, 128, 0, st>>>(xyz, nxyz, N, S, K,
                                                            rr[br] * rr[br], gbuf);
    int CinP = al32(Cp + 3);
    long long tot = (long long)M * CinP;
    bf16_t* G = (bf16_t*)P[0];
    group_gather_kernel<<<(unsigned)((tot + 255) / 256), 256, 0, st>>>(
        xyz, nxyz, points, Cp, gbuf, N, S, K, CinP, G);
    MlpOut mo = run_mlp(st, G, M, brs[br], 3, P, wp, cs, cs2);
    int C3 = brs[br][2].Cout;
    maxpool_kernel<<<(BB * S * C3 + 255) / 256, 256, 0, st>>>(mo.f, S, K, C3, out_p, Ctot, coloff);
    coloff += C3;
  }
}

static void run_fp(hipStream_t st, const float* xyz1, int N1, const float* xyz2, int N2,
                   const float* p1, int C1, const float* p2, int C2,
                   const Mlp* L, int nL, float* out_persist,
                   char* const P[3], bf16_t* wp, float* cs, float* cs2) {
  int M = BB * N1;
  int Cin = C1 + C2;            // all fp/head Cin values are multiples of 32
  int CinP = al32(Cin);
  bf16_t* CAT = (bf16_t*)P[0];
  if (p1 && C1 > 0) {
    long long tot = (long long)M * C1;
    skip_copy_kernel<<<(unsigned)((tot + 255) / 256), 256, 0, st>>>(p1, M, C1, CAT, CinP, 0);
  }
  interp_kernel<<<dim3((unsigned)N1, BB), 128, 0, st>>>(xyz1, xyz2, p2, N1, N2, C2,
                                                        CAT, CinP, C1);
  MlpOut mo = run_mlp(st, CAT, M, L, nL, P, wp, cs, cs2);
  hipMemcpyAsync(out_persist, mo.f, (size_t)M * L[nL - 1].Cout * sizeof(float),
                 hipMemcpyDeviceToDevice, st);
}

extern "C" void kernel_launch(void* const* d_in, const int* in_sizes, int n_in,
                              void* d_out, int out_size, void* d_ws, size_t ws_size,
                              hipStream_t stream) {
  (void)in_sizes; (void)n_in; (void)out_size; (void)ws_size;
  const float* x = (const float*)d_in[0];

  // params flattened in jax pytree order: dict keys sorted, layer dicts W,b,be,g
  auto L = [&](int i, int cin, int cout) {
    Mlp m;
    m.W  = (const float*)d_in[i];
    m.b  = (const float*)d_in[i + 1];
    m.be = (const float*)d_in[i + 2];
    m.g  = (const float*)d_in[i + 3];
    m.Cin = cin; m.Cout = cout;
    return m;
  };
  Mlp fp1L[2]  = {L(1, 512, 512),    L(5, 512, 512)};
  Mlp fp2L[2]  = {L(9, 1120, 1024),  L(13, 1024, 512)};
  Mlp fp3L[2]  = {L(17, 1024, 1120), L(21, 1120, 1024)};
  Mlp headL[2] = {L(25, 512, 512),   L(29, 512, 128)};
  Mlp sa1b0[3] = {L(33, 6, 16),   L(37, 16, 16),   L(41, 16, 32)};
  Mlp sa1b1[3] = {L(45, 6, 32),   L(49, 32, 32),   L(53, 32, 64)};
  Mlp sa2b0[3] = {L(57, 99, 128), L(61, 128, 128), L(65, 128, 256)};
  Mlp sa2b1[3] = {L(69, 99, 128), L(73, 128, 196), L(77, 196, 256)};
  Mlp sa3b0[3] = {L(81, 515, 128), L(85, 128, 128), L(89, 128, 256)};
  Mlp sa3b1[3] = {L(93, 515, 128), L(97, 128, 196), L(101, 196, 256)};

  char* base = (char*)d_ws;
  size_t off = 0;
  auto alloc = [&](size_t bytes) -> void* {
    void* p = base + off;
    off += (bytes + 255) & ~(size_t)255;
    return p;
  };

  // persistent buffers
  int*   fidx1 = (int*)alloc((size_t)BB * 512 * sizeof(int));
  float* l1x   = (float*)alloc((size_t)BB * 512 * 3 * sizeof(float));
  float* l1p   = (float*)alloc((size_t)BB * 512 * 96 * sizeof(float));
  int*   fidx2 = (int*)alloc((size_t)BB * 256 * sizeof(int));
  float* l2x   = (float*)alloc((size_t)BB * 256 * 3 * sizeof(float));
  float* l2p   = (float*)alloc((size_t)BB * 256 * 512 * sizeof(float));
  int*   fidx3 = (int*)alloc((size_t)BB * 128 * sizeof(int));
  float* l3x   = (float*)alloc((size_t)BB * 128 * 3 * sizeof(float));
  float* l3p   = (float*)alloc((size_t)BB * 128 * 512 * sizeof(float));
  float* l2pn  = (float*)alloc((size_t)BB * 256 * 1024 * sizeof(float));
  float* l1pn  = (float*)alloc((size_t)BB * 512 * 512 * sizeof(float));
  float* l0pn  = (float*)alloc((size_t)BB * 4096 * 512 * sizeof(float));
  float* cs    = (float*)alloc(1120 * sizeof(float));
  float* cs2   = (float*)alloc(1120 * sizeof(float));
  int*   gbuf  = (int*)alloc((size_t)131072 * sizeof(int));
  bf16_t* wp   = (bf16_t*)alloc((size_t)4 << 20);   // packed weights (max ~2.3MB)

  // 3-region rotating transient arena (max: 131072 x al32(515)=544 bf16 = 136MB)
  const size_t REG = (size_t)140 << 20;
  char* P[3] = {(char*)alloc(REG), (char*)alloc(REG), (char*)alloc(REG)};

  // SA stages (multi-scale grouping)
  run_sa(stream, x,   4096, x,   3,   512, 0.1f, 16,  sa1b0, 0.2f, 32,  sa1b1,
         fidx1, l1x, l1p, 96,  P, wp, cs, cs2, gbuf);
  run_sa(stream, l1x, 512,  l1p, 96,  256, 0.2f, 64,  sa2b0, 0.4f, 128, sa2b1,
         fidx2, l2x, l2p, 512, P, wp, cs, cs2, gbuf);
  run_sa(stream, l2x, 256,  l2p, 512, 128, 0.4f, 128, sa3b0, 0.8f, 256, sa3b1,
         fidx3, l3x, l3p, 512, P, wp, cs, cs2, gbuf);

  // FP stages (feature propagation)
  run_fp(stream, l2x, 256,  l3x, 128, l2p,     512, l3p,  512,  fp3L, 2, l2pn, P, wp, cs, cs2);
  run_fp(stream, l1x, 512,  l2x, 256, l1p,     96,  l2pn, 1024, fp2L, 2, l1pn, P, wp, cs, cs2);
  run_fp(stream, x,   4096, l1x, 512, nullptr, 0,   l1pn, 512,  fp1L, 2, l0pn, P, wp, cs, cs2);

  // head MLP + transpose to (B, 128, 4096)
  {
    int M = BB * 4096;
    bf16_t* H = (bf16_t*)P[0];
    long long tot = (long long)M * 512;
    skip_copy_kernel<<<(unsigned)((tot + 255) / 256), 256, 0, stream>>>(l0pn, M, 512, H, 512, 0);
    MlpOut mo = run_mlp(stream, H, M, headL, 2, P, wp, cs, cs2);
    long long ot = (long long)M * 128;
    transpose_out_kernel<<<(unsigned)((ot + 255) / 256), 256, 0, stream>>>(
        mo.f, 4096, 128, (float*)d_out);
  }
}
